// SelfAttention_42623255445761
// MI455X (gfx1250) — matrix operand order repaired and checked by reading, not compiled, not run on hardware
//
#include <hip/hip_runtime.h>

#define B_  4
#define L_  1024
#define D_  1024
#define H_  16
#define HD_ 64

typedef __attribute__((ext_vector_type(16))) __bf16        v16bf;
typedef __attribute__((ext_vector_type(8)))  float         v8f;
typedef __attribute__((ext_vector_type(4)))  unsigned int  v4u;
typedef __attribute__((ext_vector_type(8)))  int           v8i;
typedef __attribute__((ext_vector_type(4)))  int           v4i;

union Frag { v16bf v; uint4 q[2]; };

constexpr int BM  = 128;
constexpr int BN  = 128;
constexpr int BK  = 32;
constexpr int LDT = BK + 8;   // padded LDS row stride (bf16) -> conflict-free, 16B aligned rows

// ---------------------------------------------------------------------------
// fp32 -> bf16 elementwise prepass (activations + weights), float4 granular
// ---------------------------------------------------------------------------
__global__ __launch_bounds__(256)
void cvt_f32_bf16(const float* __restrict__ src, __bf16* __restrict__ dst, int n4)
{
    const int i = blockIdx.x * blockDim.x + threadIdx.x;
    if (i < n4) {
        const float4 f = ((const float4*)src)[i];
        alignas(8) __bf16 t[4] = { (__bf16)f.x, (__bf16)f.y, (__bf16)f.z, (__bf16)f.w };
        ((uint2*)dst)[i] = *(const uint2*)t;
    }
}

// ---------------------------------------------------------------------------
// TDM: DMA a 2D bf16 tile (tile_d0 x tile_d1 elements, row stride stride_d0)
// into LDS at lds_byte_addr, inserting 4 DWORDs of padding after every
// 16 DWORDs (64B row) -> LDS row stride = 40 bf16 = LDT.
// ---------------------------------------------------------------------------
__device__ __forceinline__ void tdm_load_tile_2d(unsigned lds_byte_addr,
                                                 const __bf16* gsrc,
                                                 unsigned tensor_d0, unsigned tensor_d1,
                                                 unsigned stride_d0,
                                                 unsigned tile_d0, unsigned tile_d1)
{
    const unsigned long long ga = (unsigned long long)(size_t)gsrc;
    v4u g0;
    g0[0] = 1u;                                              // count=1, user descriptor
    g0[1] = lds_byte_addr;                                   // LDS byte address
    g0[2] = (unsigned)(ga & 0xFFFFFFFFu);                    // global addr [31:0]
    g0[3] = (unsigned)((ga >> 32) & 0x01FFFFFFu)             // global addr [56:32]
          | (2u << 30);                                      // type = 2 ("image")
    v8i g1;
    g1[0] = (int)( (1u << 16)        // data_size = 1 -> 2 bytes
                 | (1u << 20)        // pad_enable
                 | (3u << 22)        // pad_interval = 3 -> every 16 DWORDs (64B)
                 | (3u << 25) );     // pad_amount   = 3 -> 4 DWORDs (16B)
    g1[1] = (int)((tensor_d0 & 0xFFFFu) << 16);              // tensor_dim0 low 16 -> d1[31:16]
    g1[2] = (int)(((tensor_d0 >> 16) & 0xFFFFu)              // tensor_dim0 high 16
                 | ((tensor_d1 & 0xFFFFu) << 16));           // tensor_dim1 low 16
    g1[3] = (int)(((tensor_d1 >> 16) & 0xFFFFu)              // tensor_dim1 high 16
                 | ((tile_d0 & 0xFFFFu) << 16));             // tile_dim0
    g1[4] = (int)(tile_d1 & 0xFFFFu);                        // tile_dim1 (tile_dim2 = 0)
    g1[5] = (int)stride_d0;                                  // tensor_dim0_stride [31:0]
    g1[6] = 0;                                               // stride high bits / dim1_stride low
    g1[7] = 0;
    const v4i z4 = { 0, 0, 0, 0 };
    const v8i z8 = { 0, 0, 0, 0, 0, 0, 0, 0 };
    __builtin_amdgcn_tensor_load_to_lds(g0, g1, z4, z4, z8, 0);
}

// ---------------------------------------------------------------------------
// GEMM: out = A @ W^T + bias    (A: MxK bf16, W: NxK bf16 row-major)
// mode 0: store bf16 head-major [B,H,L,hd]            (Q, K projections)
// mode 1: store bf16 head-major transposed [B,H,hd,L] (V projection)
// mode 2: store fp32 row-major [M,N]                  (output projection)
// ---------------------------------------------------------------------------
__global__ __launch_bounds__(256)
void proj_gemm(const __bf16* __restrict__ A, const __bf16* __restrict__ W,
               const float* __restrict__ bias,
               __bf16* __restrict__ out_bf16, float* __restrict__ out_f32,
               int M, int N, int K, int mode)
{
    __shared__ __bf16 As[BM * LDT];
    __shared__ __bf16 Bs[BN * LDT];

    const int tid   = threadIdx.x;
    const int lane  = tid & 31;
    const int wid   = tid >> 5;       // 0..7
    const int waveM = wid & 3;        // 4 waves along M
    const int waveN = wid >> 2;       // 2 waves along N
    const int m0 = blockIdx.y * BM;
    const int n0 = blockIdx.x * BN;

    const unsigned asAddr = (unsigned)(size_t)(void*)&As[0];   // low 32 bits = LDS offset
    const unsigned bsAddr = (unsigned)(size_t)(void*)&Bs[0];

    v8f acc[2][4] = {};

    const int lr    = lane & 15;
    const int halfA = (lane < 16) ? 0 : 8;    // A-fragment K-half offset
    const int halfB = (lane < 16) ? 0 : 8;    // B-fragment K-half offset

    for (int k0 = 0; k0 < K; k0 += BK) {
        // ---- stage both tiles via the Tensor Data Mover (wave 0 only) ----
        if (wid == 0) {
            tdm_load_tile_2d(asAddr, A + (size_t)m0 * K + k0,
                             (unsigned)K, (unsigned)M, (unsigned)K, BK, BM);
            tdm_load_tile_2d(bsAddr, W + (size_t)n0 * K + k0,
                             (unsigned)K, (unsigned)N, (unsigned)K, BK, BN);
            __builtin_amdgcn_s_wait_tensorcnt(0);
        }
        __syncthreads();

        // ---- fragments + 8 WMMAs per K-step ----
        Frag a[2], b[4];
        #pragma unroll
        for (int mi = 0; mi < 2; ++mi) {
            const __bf16* p = &As[(waveM * 32 + mi * 16 + lr) * LDT + halfA];
            a[mi].q[0] = *(const uint4*)p;
            a[mi].q[1] = *(const uint4*)(p + 16);
        }
        #pragma unroll
        for (int ni = 0; ni < 4; ++ni) {
            const __bf16* p = &Bs[(waveN * 64 + ni * 16 + lr) * LDT + halfB];
            b[ni].q[0] = *(const uint4*)p;
            b[ni].q[1] = *(const uint4*)(p + 16);
        }
        #pragma unroll
        for (int mi = 0; mi < 2; ++mi)
            #pragma unroll
            for (int ni = 0; ni < 4; ++ni)
                acc[mi][ni] = __builtin_amdgcn_wmma_f32_16x16x32_bf16(
                    false, a[mi].v, false, b[ni].v, (short)0, acc[mi][ni], false, false);
        __syncthreads();
    }

    // ---- epilogue ----
    const int rowHalf = (lane >> 4) * 8;   // 0 or 8
    #pragma unroll
    for (int mi = 0; mi < 2; ++mi) {
        #pragma unroll
        for (int ni = 0; ni < 4; ++ni) {
            const int gn = n0 + waveN * 64 + ni * 16 + lr;
            const float bv = bias[gn];
            #pragma unroll
            for (int r = 0; r < 8; ++r) {
                const int gm = m0 + waveM * 32 + mi * 16 + r + rowHalf;
                const float val = acc[mi][ni][r] + bv;
                if (mode == 2) {
                    out_f32[(size_t)gm * N + gn] = val;
                } else {
                    const int bb = gm >> 10;         // / L
                    const int lp = gm & (L_ - 1);
                    const int hh = gn >> 6;          // / HD
                    const int dd = gn & (HD_ - 1);
                    size_t idx;
                    if (mode == 0) idx = (((size_t)(bb * H_ + hh)) * L_ + lp) * HD_ + dd;
                    else           idx = (((size_t)(bb * H_ + hh)) * HD_ + dd) * L_ + lp;
                    out_bf16[idx] = (__bf16)val;
                }
            }
        }
    }
}

// ---------------------------------------------------------------------------
// Flash-attention: 1 wave = 16 query rows, online softmax over 32-key blocks.
// Q,K: [B,H,L,hd] bf16; Vt: [B,H,hd,L] bf16; Y: [B,L,D] bf16
// ---------------------------------------------------------------------------
__global__ __launch_bounds__(128)
void attn_kernel(const __bf16* __restrict__ Q, const __bf16* __restrict__ Kb,
                 const __bf16* __restrict__ Vt, __bf16* __restrict__ Y)
{
    __shared__ __bf16 Pl[4 * 16 * LDT];    // per-wave P-tile staging (S^T relayout)

    const int lane = threadIdx.x & 31;
    const int wid  = threadIdx.x >> 5;     // 0..3
    const int bh   = blockIdx.y;           // b*H + h
    const int q0   = blockIdx.x * 64 + wid * 16;
    const int b    = bh >> 4;
    const int h    = bh & 15;

    const __bf16* Qh = Q  + (size_t)bh * L_ * HD_;
    const __bf16* Kh = Kb + (size_t)bh * L_ * HD_;
    const __bf16* Vh = Vt + (size_t)bh * HD_ * L_;
    __bf16* Pw = &Pl[wid * 16 * LDT];

    const int lr    = lane & 15;
    const int half8 = (lane >> 4) * 8;
    const int halfA = (lane < 16) ? 0 : 8;
    const int halfB = (lane < 16) ? 0 : 8;

    // Q A-fragments: 2 k-steps of 32 over hd=64
    Frag qf[2];
    #pragma unroll
    for (int s = 0; s < 2; ++s) {
        const __bf16* p = Qh + (size_t)(q0 + lr) * HD_ + s * 32 + halfA;
        qf[s].q[0] = *(const uint4*)p;
        qf[s].q[1] = *(const uint4*)(p + 16);
    }

    float mrow[8], lsum[8];
    #pragma unroll
    for (int r = 0; r < 8; ++r) { mrow[r] = -1e30f; lsum[r] = 0.f; }
    v8f o[4] = {};

    const float SCALE = 0.125f;  // 1/sqrt(64)
    const int nkb = ((q0 + 15) >> 5) + 1;

    for (int kb32 = 0; kb32 < nkb; ++kb32) {
        const int kb = kb32 * 32;

        // K B-fragments: 2 key groups x 2 d-steps
        Frag kf[2][2];
        #pragma unroll
        for (int g = 0; g < 2; ++g)
            #pragma unroll
            for (int s = 0; s < 2; ++s) {
                const __bf16* p = Kh + (size_t)(kb + g * 16 + lr) * HD_ + s * 32 + halfB;
                kf[g][s].q[0] = *(const uint4*)p;
                kf[g][s].q[1] = *(const uint4*)(p + 16);
            }

        v8f s0 = {}, s1 = {};
        s0 = __builtin_amdgcn_wmma_f32_16x16x32_bf16(false, qf[0].v, false, kf[0][0].v, (short)0, s0, false, false);
        s0 = __builtin_amdgcn_wmma_f32_16x16x32_bf16(false, qf[1].v, false, kf[0][1].v, (short)0, s0, false, false);
        s1 = __builtin_amdgcn_wmma_f32_16x16x32_bf16(false, qf[0].v, false, kf[1][0].v, (short)0, s1, false, false);
        s1 = __builtin_amdgcn_wmma_f32_16x16x32_bf16(false, qf[1].v, false, kf[1][1].v, (short)0, s1, false, false);

        // scale, causal mask, online softmax (rows live in 16-lane halves)
        float p0v[8], p1v[8], alpha[8];
        #pragma unroll
        for (int r = 0; r < 8; ++r) {
            const int qg = q0 + r + half8;
            const float x0 = (kb + lr      <= qg) ? s0[r] * SCALE : -1e30f;
            const float x1 = (kb + 16 + lr <= qg) ? s1[r] * SCALE : -1e30f;
            float mx = fmaxf(x0, x1);
            mx = fmaxf(mx, __shfl_xor(mx, 1));
            mx = fmaxf(mx, __shfl_xor(mx, 2));
            mx = fmaxf(mx, __shfl_xor(mx, 4));
            mx = fmaxf(mx, __shfl_xor(mx, 8));
            const float mnew = fmaxf(mrow[r], mx);
            const float al = __expf(mrow[r] - mnew);
            const float e0 = __expf(x0 - mnew);
            const float e1 = __expf(x1 - mnew);
            float rs = e0 + e1;
            rs += __shfl_xor(rs, 1);
            rs += __shfl_xor(rs, 2);
            rs += __shfl_xor(rs, 4);
            rs += __shfl_xor(rs, 8);
            lsum[r] = lsum[r] * al + rs;
            mrow[r] = mnew;
            alpha[r] = al;
            p0v[r] = e0;
            p1v[r] = e1;
        }

        #pragma unroll
        for (int nt = 0; nt < 4; ++nt)
            #pragma unroll
            for (int r = 0; r < 8; ++r) o[nt][r] *= alpha[r];

        // relayout P (C/D layout -> A-fragment layout) via per-wave LDS
        #pragma unroll
        for (int r = 0; r < 8; ++r) {
            Pw[(r + half8) * LDT + lr]      = (__bf16)p0v[r];
            Pw[(r + half8) * LDT + 16 + lr] = (__bf16)p1v[r];
        }
        asm volatile("s_wait_dscnt 0" ::: "memory");
        Frag pf;
        {
            const __bf16* p = &Pw[lr * LDT + halfA];
            pf.q[0] = *(const uint4*)p;
            pf.q[1] = *(const uint4*)(p + 16);
        }

        // O += P @ V   (V^T rows are contiguous in keys)
        #pragma unroll
        for (int nt = 0; nt < 4; ++nt) {
            Frag vf;
            const __bf16* p = Vh + (size_t)(nt * 16 + lr) * L_ + kb + halfB;
            vf.q[0] = *(const uint4*)p;
            vf.q[1] = *(const uint4*)(p + 16);
            o[nt] = __builtin_amdgcn_wmma_f32_16x16x32_bf16(
                false, pf.v, false, vf.v, (short)0, o[nt], false, false);
        }
    }

    // normalize + store Y [B,L,D] bf16
    float inv[8];
    #pragma unroll
    for (int r = 0; r < 8; ++r) inv[r] = 1.f / lsum[r];
    #pragma unroll
    for (int nt = 0; nt < 4; ++nt)
        #pragma unroll
        for (int r = 0; r < 8; ++r) {
            const int qg = q0 + r + half8;
            const int n  = h * 64 + nt * 16 + lr;
            Y[((size_t)(b * L_ + qg)) * D_ + n] = (__bf16)(o[nt][r] * inv[r]);
        }
}

// ---------------------------------------------------------------------------
extern "C" void kernel_launch(void* const* d_in, const int* in_sizes, int n_in,
                              void* d_out, int out_size, void* d_ws, size_t ws_size,
                              hipStream_t stream)
{
    const float* key   = (const float*)d_in[0];
    const float* value = (const float*)d_in[1];
    const float* query = (const float*)d_in[2];
    const float* Wk = (const float*)d_in[3];
    const float* bk = (const float*)d_in[4];
    const float* Wq = (const float*)d_in[5];
    const float* bq = (const float*)d_in[6];
    const float* Wv = (const float*)d_in[7];
    const float* bv = (const float*)d_in[8];
    const float* Wp = (const float*)d_in[9];
    const float* bp = (const float*)d_in[10];

    char* ws = (char*)d_ws;
    const size_t ACT = (size_t)B_ * L_ * D_ * sizeof(__bf16);   // 8 MB
    const size_t WSZ = (size_t)D_ * D_ * sizeof(__bf16);        // 2 MB
    __bf16* Xq  = (__bf16*)(ws);
    __bf16* Xk  = (__bf16*)(ws + ACT);
    __bf16* Xv  = (__bf16*)(ws + 2 * ACT);
    __bf16* Wqb = (__bf16*)(ws + 3 * ACT);
    __bf16* Wkb = (__bf16*)(ws + 3 * ACT + WSZ);
    __bf16* Wvb = (__bf16*)(ws + 3 * ACT + 2 * WSZ);
    __bf16* Wpb = (__bf16*)(ws + 3 * ACT + 3 * WSZ);
    __bf16* Qb  = (__bf16*)(ws + 3 * ACT + 4 * WSZ);
    __bf16* Kbf = (__bf16*)(ws + 4 * ACT + 4 * WSZ);
    __bf16* Vbf = (__bf16*)(ws + 5 * ACT + 4 * WSZ);
    __bf16* Yb  = (__bf16*)(ws + 6 * ACT + 4 * WSZ);            // total 64 MB

    // --- fp32 -> bf16 prepass ---
    const int nAct4 = (B_ * L_ * D_) / 4;   // 1M float4
    const int nW4   = (D_ * D_) / 4;        // 256K float4
    cvt_f32_bf16<<<dim3((nAct4 + 255) / 256), dim3(256), 0, stream>>>(query, Xq, nAct4);
    cvt_f32_bf16<<<dim3((nAct4 + 255) / 256), dim3(256), 0, stream>>>(key,   Xk, nAct4);
    cvt_f32_bf16<<<dim3((nAct4 + 255) / 256), dim3(256), 0, stream>>>(value, Xv, nAct4);
    cvt_f32_bf16<<<dim3((nW4 + 255) / 256),   dim3(256), 0, stream>>>(Wq, Wqb, nW4);
    cvt_f32_bf16<<<dim3((nW4 + 255) / 256),   dim3(256), 0, stream>>>(Wk, Wkb, nW4);
    cvt_f32_bf16<<<dim3((nW4 + 255) / 256),   dim3(256), 0, stream>>>(Wv, Wvb, nW4);
    cvt_f32_bf16<<<dim3((nW4 + 255) / 256),   dim3(256), 0, stream>>>(Wp, Wpb, nW4);

    // --- projections (TDM-staged WMMA GEMMs) ---
    dim3 gblk(D_ / BN, (B_ * L_) / BM);   // (8, 32)
    dim3 tblk(256);
    proj_gemm<<<gblk, tblk, 0, stream>>>(Xq, Wqb, bq, Qb,  nullptr, B_ * L_, D_, D_, 0);
    proj_gemm<<<gblk, tblk, 0, stream>>>(Xk, Wkb, bk, Kbf, nullptr, B_ * L_, D_, D_, 0);
    proj_gemm<<<gblk, tblk, 0, stream>>>(Xv, Wvb, bv, Vbf, nullptr, B_ * L_, D_, D_, 1);

    // --- flash attention ---
    dim3 gattn(L_ / 64, B_ * H_);
    attn_kernel<<<gattn, dim3(128), 0, stream>>>(Qb, Kbf, Vbf, Yb);

    // --- output projection ---
    proj_gemm<<<gblk, tblk, 0, stream>>>(Yb, Wpb, bp, nullptr, (float*)d_out, B_ * L_, D_, D_, 2);
}
